// TransformerBlock_ema_75582834475260
// MI455X (gfx1250) — matrix operand order repaired and checked by reading, not compiled, not run on hardware
//
#include <hip/hip_runtime.h>

// ---------------------------------------------------------------------------
// Types for WMMA fragments / TDM descriptors (wave32, gfx1250)
// ---------------------------------------------------------------------------
typedef __attribute__((ext_vector_type(16))) _Float16 v16h;
typedef __attribute__((ext_vector_type(8)))  _Float16 v8h;
typedef __attribute__((ext_vector_type(8)))  float    v8f;
typedef __attribute__((ext_vector_type(4)))  unsigned int v4u;
typedef __attribute__((ext_vector_type(8)))  int      v8i;
typedef __attribute__((ext_vector_type(4)))  int      v4i;

union F16x16 { v16h v; v8h h[2]; };

#if __has_builtin(__builtin_amdgcn_tensor_load_to_lds) && __has_builtin(__builtin_amdgcn_s_wait_tensorcnt)
#define HAVE_TDM 1
#else
#define HAVE_TDM 0
#endif

// Problem constants (match setup_inputs)
#define BB   8
#define CC   256
#define HWP  4096          // 64*64
#define NH   8             // heads H
#define NG   8             // groups G
#define HID  680           // int(256*2.66)
#define CH2  1360          // 2*HID
#define KP   704           // HID padded to multiple of 64

// ---------------------------------------------------------------------------
// Stage the 16xK f16 A-tile for this block into LDS.
// TDM path: wave 0 issues one tensor_load_to_lds (2D descriptor: element=2B,
// tensor_dim0=K, dim0_stride=K, tile 16 rows x K cols, lds_addr=0 since sA is
// the block's only LDS allocation), waits TENSORcnt, then block barrier.
// The opaque asm afterwards tells the compiler the DMA wrote sA (otherwise
// the ds_loads of a "store-free" shared array get folded to undef).
// ---------------------------------------------------------------------------
__device__ __forceinline__ void stage_A_tile(const _Float16* __restrict__ A,
                                             int m0, int K, int Mtot,
                                             _Float16* sA, int tid) {
#if HAVE_TDM
  if (tid < 32) {  // one wave issues the DMA (TDM ignores EXEC; per-wave issue)
    unsigned long long ga = (unsigned long long)(size_t)(A + (size_t)m0 * K);
    v4u g0 = { 1u,                                    // count=1, is_restore=0
               0u,                                    // lds_addr = 0 (sA base)
               (unsigned)(ga & 0xFFFFFFFFu),          // global_addr[31:0]
               (unsigned)((ga >> 32) & 0x01FFFFFFu) | (2u << 30) };  // type=2
    v8i g1 = { (int)(1u << 16),                       // data_size=1 (2 bytes)
               (int)(((unsigned)K & 0xFFFFu) << 16),  // tensor_dim0[15:0]
               (int)((((unsigned)K >> 16) & 0xFFFFu) |
                     (((unsigned)Mtot & 0xFFFFu) << 16)),  // dim0 hi | dim1 lo
               (int)((((unsigned)Mtot >> 16) & 0xFFFFu) |
                     (((unsigned)K & 0xFFFFu) << 16)),     // dim1 hi | tile_dim0=K
               16,                                    // tile_dim1 = 16 rows
               K,                                     // tensor_dim0_stride[31:0]
               0, 0 };
    v4i gz4 = { 0, 0, 0, 0 };
    v8i gz8 = { 0, 0, 0, 0, 0, 0, 0, 0 };
    __builtin_amdgcn_tensor_load_to_lds(g0, g1, gz4, gz4, gz8, 0);
    __builtin_amdgcn_s_wait_tensorcnt((short)0);
  }
  // Make the TDM's LDS write visible to the optimizer: opaque use of sA with
  // a memory clobber so subsequent ds_loads are not folded away.
  asm volatile("" : : "v"(sA) : "memory");
  __syncthreads();
#else
  for (int i = tid; i < 16 * K; i += 128) sA[i] = A[(size_t)m0 * K + i];
  __syncthreads();
#endif
}

// Load 4 B fragments (16x32 K-slabs for 4 adjacent N-tiles) from pixel-major B
__device__ __forceinline__ void load_B4(const _Float16* const* Brow, int k0,
                                        v16h* bv) {
#pragma unroll
  for (int t = 0; t < 4; ++t) bv[t] = *(const v16h*)(Brow[t] + k0);
}

// A fragment (16x32) from LDS at k0 for this lane
__device__ __forceinline__ v16h load_A_lds(const _Float16* sA, int K, int lane,
                                           int k0) {
  const _Float16* pa = sA + (lane & 15) * K + ((lane >> 4) << 4) + k0;
  F16x16 a;
  a.h[0] = *(const v8h*)(pa);
  a.h[1] = *(const v8h*)(pa + 8);
  return a.v;
}

// ---------------------------------------------------------------------------
// Kernel 0: convert weights to f16 (w_out zero-padded 680 -> 704 cols)
// ---------------------------------------------------------------------------
__global__ __launch_bounds__(256) void k_prep(const float* __restrict__ win,
                                              const float* __restrict__ wout,
                                              _Float16* __restrict__ winh,
                                              _Float16* __restrict__ wouth) {
  int i = blockIdx.x * 256 + threadIdx.x;
  if (i < CH2 * CC) winh[i] = (_Float16)win[i];
  if (i < CC * KP) {
    int r = i / KP, c = i % KP;
    wouth[i] = (c < HID) ? (_Float16)wout[r * HID + c] : (_Float16)0.f;
  }
}

// ---------------------------------------------------------------------------
// Channel LayerNorm over 256 channels per pixel.
//   y32  (optional): fp32, same [b,c,p] layout as x
//   y16t (optional): f16, pixel-major [b, p, c] (GEMM B-operand layout)
// ---------------------------------------------------------------------------
__global__ __launch_bounds__(256) void k_ln(const float* __restrict__ x,
                                            const float* __restrict__ w,
                                            const float* __restrict__ bi,
                                            float* __restrict__ y32,
                                            _Float16* __restrict__ y16t) {
  int tid = threadIdx.x, px = tid & 63, q = tid >> 6;
  int p = blockIdx.x * 64 + px;
  size_t xb = (size_t)blockIdx.z * CC * HWP;
  __shared__ float rs[256], rq[256], smu[64], sin[64];
  float s = 0.f, ss = 0.f;
  for (int c = q * 64; c < q * 64 + 64; ++c) {
    float v = x[xb + (size_t)c * HWP + p];
    s += v; ss += v * v;
  }
  rs[tid] = s; rq[tid] = ss;
  __syncthreads();
  if (q == 0) {
    float ts = rs[px] + rs[px + 64] + rs[px + 128] + rs[px + 192];
    float tq = rq[px] + rq[px + 64] + rq[px + 128] + rq[px + 192];
    float mu = ts * (1.f / CC);
    float var = tq * (1.f / CC) - mu * mu;
    smu[px] = mu; sin[px] = rsqrtf(var + 1e-5f);
  }
  __syncthreads();
  float mu = smu[px], inv = sin[px];
  for (int c = q * 64; c < q * 64 + 64; ++c) {
    float v = (x[xb + (size_t)c * HWP + p] - mu) * inv * w[c] + bi[c];
    if (y32)  y32[xb + (size_t)c * HWP + p] = v;
    if (y16t) y16t[(size_t)blockIdx.z * HWP * CC + (size_t)p * CC + c] = (_Float16)v;
  }
}

// ---------------------------------------------------------------------------
// attn_ema fused: one workgroup per (b, g, n) = 4 channels x 64x64 pixels.
// ---------------------------------------------------------------------------
__device__ __forceinline__ float conv_at(const _Float16* sG, const float* sW3,
                                         const float* sB3, int o, int i, int j) {
  float v = sB3[o];
  const float* wk = sW3 + o * 36;
#pragma unroll
  for (int ci = 0; ci < 4; ++ci) {
    const float* w9 = wk + ci * 9;
    const _Float16* gch = sG + ci * HWP;
#pragma unroll
    for (int dy = -1; dy <= 1; ++dy) {
      int ii = i + dy; if (ii < 0 || ii > 63) continue;
#pragma unroll
      for (int dx = -1; dx <= 1; ++dx) {
        int jj = j + dx; if (jj < 0 || jj > 63) continue;
        v += w9[(dy + 1) * 3 + (dx + 1)] * (float)gch[(ii << 6) + jj];
      }
    }
  }
  return v;
}

__global__ __launch_bounds__(256) void k_attn(const float* __restrict__ y1,
                                              const float* __restrict__ x,
                                              const float* __restrict__ w1,
                                              const float* __restrict__ b1,
                                              const float* __restrict__ w3,
                                              const float* __restrict__ b3,
                                              const float* __restrict__ gnw,
                                              const float* __restrict__ gnb,
                                              float* __restrict__ xmid) {
  int gi = blockIdx.x, n = blockIdx.y, bb = blockIdx.z;
  int tid = threadIdx.x;
  __shared__ _Float16 sG[4 * HWP];
  __shared__ float sxh[256], sxw[256], ssh[256], ssw[256];
  __shared__ float red[256];
  __shared__ float sW3[144], sB3[4], smu[4], sinv[4], sa1[4], sa2[4], sm2[4];

  int chBase = n * 32 + gi * 4;
  size_t gOff = ((size_t)bb * CC + chBase) * HWP;

  if (tid < 144) sW3[tid] = w3[n * 144 + tid];
  if (tid < 4)   sB3[tid] = b3[n * 4 + tid];

  int c = tid >> 6, l = tid & 63;
  for (int k2 = 0; k2 < 64; ++k2) {
    int p = l + (k2 << 6);
    sG[c * HWP + p] = (_Float16)y1[gOff + (size_t)c * HWP + p];
  }
  __syncthreads();
  {
    float s = 0.f;
    for (int j = 0; j < 64; ++j) s += (float)sG[c * HWP + (l << 6) + j];
    sxh[c * 64 + l] = s * (1.f / 64);
    float s2 = 0.f;
    for (int i = 0; i < 64; ++i) s2 += (float)sG[c * HWP + (i << 6) + l];
    sxw[c * 64 + l] = s2 * (1.f / 64);
  }
  __syncthreads();
  {
    float sh = b1[n * 4 + c], sw = sh;
#pragma unroll
    for (int i = 0; i < 4; ++i) {
      float wv = w1[(n * 4 + c) * 4 + i];
      sh += wv * sxh[i * 64 + l];
      sw += wv * sxw[i * 64 + l];
    }
    ssh[c * 64 + l] = 1.f / (1.f + expf(-sh));
    ssw[c * 64 + l] = 1.f / (1.f + expf(-sw));
  }
  __syncthreads();
  float s = 0.f, ss2 = 0.f;
  for (int k2 = 0; k2 < 64; ++k2) {
    int p = l + (k2 << 6);
    float gate = (float)sG[c * HWP + p] * ssh[c * 64 + k2] * ssw[c * 64 + l];
    s += gate; ss2 += gate * gate;
  }
  red[tid] = s;
  for (int off = 32; off; off >>= 1) { __syncthreads(); if ((tid & 63) < off) red[tid] += red[tid + off]; }
  __syncthreads();
  if (l == 0) smu[c] = red[c << 6] * (1.f / HWP);
  __syncthreads();
  red[tid] = ss2;
  for (int off = 32; off; off >>= 1) { __syncthreads(); if ((tid & 63) < off) red[tid] += red[tid + off]; }
  __syncthreads();
  if (l == 0) {
    float mu = smu[c];
    float var = red[c << 6] * (1.f / HWP) - mu * mu;
    sinv[c] = rsqrtf(var + 1e-5f);
  }
  __syncthreads();
  float pm[4] = {0.f, 0.f, 0.f, 0.f};
  for (int k2 = 0; k2 < 16; ++k2) {
    int p = tid + (k2 << 8), i = p >> 6, j = p & 63;
#pragma unroll
    for (int o = 0; o < 4; ++o) pm[o] += conv_at(sG, sW3, sB3, o, i, j);
  }
  for (int o = 0; o < 4; ++o) {
    __syncthreads();
    red[tid] = pm[o];
    for (int off = 128; off; off >>= 1) { __syncthreads(); if (tid < off) red[tid] += red[tid + off]; }
    __syncthreads();
    if (tid == 0) sm2[o] = red[0] * (1.f / HWP);
  }
  __syncthreads();
  if (tid == 0) {
    float m1[4], m2[4], mx1 = -1e30f, mx2 = -1e30f;
    for (int c2 = 0; c2 < 4; ++c2) {
      m1[c2] = gnb[n * 4 + c2]; m2[c2] = sm2[c2];
      mx1 = fmaxf(mx1, m1[c2]); mx2 = fmaxf(mx2, m2[c2]);
    }
    float s1 = 0.f, s2 = 0.f;
    for (int c2 = 0; c2 < 4; ++c2) {
      m1[c2] = expf(m1[c2] - mx1); s1 += m1[c2];
      m2[c2] = expf(m2[c2] - mx2); s2 += m2[c2];
    }
    for (int c2 = 0; c2 < 4; ++c2) { sa1[c2] = m1[c2] / s1; sa2[c2] = m2[c2] / s2; }
  }
  __syncthreads();
  float gw[4], gb[4];
#pragma unroll
  for (int c2 = 0; c2 < 4; ++c2) { gw[c2] = gnw[n * 4 + c2]; gb[c2] = gnb[n * 4 + c2]; }
  for (int k2 = 0; k2 < 16; ++k2) {
    int p = tid + (k2 << 8), i = p >> 6, j = p & 63;
    float wm = 0.f;
#pragma unroll
    for (int c2 = 0; c2 < 4; ++c2) {
      float gv = (float)sG[c2 * HWP + p];
      float gate = gv * ssh[c2 * 64 + i] * ssw[c2 * 64 + j];
      float x1v = (gate - smu[c2]) * sinv[c2] * gw[c2] + gb[c2];
      float x2v = conv_at(sG, sW3, sB3, c2, i, j);
      wm += sa1[c2] * x2v + sa2[c2] * x1v;
    }
    float sig = 1.f / (1.f + expf(-wm));
#pragma unroll
    for (int c2 = 0; c2 < 4; ++c2)
      xmid[gOff + (size_t)c2 * HWP + p] =
          x[gOff + (size_t)c2 * HWP + p] + (float)sG[c2 * HWP + p] * sig;
  }
}

// ---------------------------------------------------------------------------
// WMMA GEMM1: Z[b, m, p] = W_in_h[m, k] * Y2t[b, p, k]   (f16 in, f16 out)
// A tile staged via TDM into LDS; B double-buffered in registers.
// grid (4096/256, 1360/16, B); block 128 (4 waves); wave -> 16x64 tile.
// ---------------------------------------------------------------------------
__global__ __launch_bounds__(128) void k_gemm1(const _Float16* __restrict__ A,
                                               const _Float16* __restrict__ Bt,
                                               _Float16* __restrict__ Z) {
  const int K = CC;
  __shared__ _Float16 sA[16 * CC];
  int tid = threadIdx.x, lane = tid & 31, wid = tid >> 5;
  int m0 = blockIdx.y << 4;
  int nb = (blockIdx.x << 8) + (wid << 6);

  stage_A_tile(A, m0, K, CH2, sA, tid);

  const _Float16* Bb = Bt + (size_t)blockIdx.z * HWP * K + ((lane >> 4) << 4);
  const _Float16* Brow[4];
#pragma unroll
  for (int t = 0; t < 4; ++t)
    Brow[t] = Bb + (size_t)(nb + (t << 4) + (lane & 15)) * K;

  v8f acc[4] = {};
  v16h b0[4], b1[4];
  load_B4(Brow, 0, b0);
  for (int k0 = 0; k0 < K; k0 += 64) {
    load_B4(Brow, k0 + 32, b1);
    __builtin_prefetch(Brow[0] + k0 + 64, 0, 0);
    v16h a0 = load_A_lds(sA, K, lane, k0);
#pragma unroll
    for (int t = 0; t < 4; ++t)
      acc[t] = __builtin_amdgcn_wmma_f32_16x16x32_f16(
          false, a0, false, b0[t], (short)0, acc[t], false, false);
    if (k0 + 64 < K) load_B4(Brow, k0 + 64, b0);
    v16h a1 = load_A_lds(sA, K, lane, k0 + 32);
#pragma unroll
    for (int t = 0; t < 4; ++t)
      acc[t] = __builtin_amdgcn_wmma_f32_16x16x32_f16(
          false, a1, false, b1[t], (short)0, acc[t], false, false);
  }

  size_t zb = (size_t)blockIdx.z * CH2 * HWP;
  int mr = m0 + ((lane >> 4) << 3);
#pragma unroll
  for (int t = 0; t < 4; ++t) {
    int nn = nb + (t << 4) + (lane & 15);
#pragma unroll
    for (int r = 0; r < 8; ++r)
      Z[zb + (size_t)(mr + r) * HWP + nn] = (_Float16)acc[t][r];
  }
}

// ---------------------------------------------------------------------------
// Depthwise 3x3 conv on 1360 ch + GELU(z1)*z2 gate; writes pixel-major f16
// [b, p, kp] padded to KP=704 (channels >= HID zero-filled).
// ---------------------------------------------------------------------------
__global__ __launch_bounds__(256) void k_dw(const _Float16* __restrict__ Z,
                                            const float* __restrict__ wdw,
                                            _Float16* __restrict__ Zg) {
  int cout = blockIdx.x, bb = blockIdx.z, tid = threadIdx.x;
  size_t zgBase = (size_t)bb * HWP * KP;
  if (cout >= HID) {
    for (int p = tid; p < HWP; p += 256)
      Zg[zgBase + (size_t)p * KP + cout] = (_Float16)0.f;
    return;
  }
  __shared__ _Float16 t1[HWP], t2[HWP];
  size_t zb = (size_t)bb * CH2 * HWP;
  for (int p = tid; p < HWP; p += 256) {
    t1[p] = Z[zb + (size_t)cout * HWP + p];
    t2[p] = Z[zb + (size_t)(cout + HID) * HWP + p];
  }
  float k1[9], k2[9];
#pragma unroll
  for (int t = 0; t < 9; ++t) {
    k1[t] = wdw[cout * 9 + t];
    k2[t] = wdw[(cout + HID) * 9 + t];
  }
  __syncthreads();
  for (int kk = 0; kk < 16; ++kk) {
    int p = tid + (kk << 8), i = p >> 6, j = p & 63;
    float a = 0.f, b2 = 0.f;
#pragma unroll
    for (int dy = -1; dy <= 1; ++dy) {
      int ii = i + dy; if (ii < 0 || ii > 63) continue;
#pragma unroll
      for (int dx = -1; dx <= 1; ++dx) {
        int jj = j + dx; if (jj < 0 || jj > 63) continue;
        int q = (ii << 6) + jj, tap = (dy + 1) * 3 + (dx + 1);
        a  += k1[tap] * (float)t1[q];
        b2 += k2[tap] * (float)t2[q];
      }
    }
    float g = 0.5f * a * (1.f + erff(a * 0.70710678f));
    Zg[zgBase + (size_t)p * KP + cout] = (_Float16)(g * b2);
  }
}

// ---------------------------------------------------------------------------
// WMMA GEMM2 + residual: out[b,m,p] = xmid[b,m,p] + W_out_h[m,k]*Zg[b,p,k]
// grid (4096/256, 256/16, B); block 128.
// ---------------------------------------------------------------------------
__global__ __launch_bounds__(128) void k_gemm2(const _Float16* __restrict__ A,
                                               const _Float16* __restrict__ Bt,
                                               const float* __restrict__ xmid,
                                               float* __restrict__ out) {
  const int K = KP;
  __shared__ _Float16 sA[16 * KP];
  int tid = threadIdx.x, lane = tid & 31, wid = tid >> 5;
  int m0 = blockIdx.y << 4;
  int nb = (blockIdx.x << 8) + (wid << 6);

  stage_A_tile(A, m0, K, CC, sA, tid);

  const _Float16* Bb = Bt + (size_t)blockIdx.z * HWP * K + ((lane >> 4) << 4);
  const _Float16* Brow[4];
#pragma unroll
  for (int t = 0; t < 4; ++t)
    Brow[t] = Bb + (size_t)(nb + (t << 4) + (lane & 15)) * K;

  v8f acc[4] = {};
  v16h b0[4], b1[4];
  load_B4(Brow, 0, b0);
  for (int k0 = 0; k0 < K; k0 += 64) {
    load_B4(Brow, k0 + 32, b1);
    __builtin_prefetch(Brow[0] + k0 + 64, 0, 0);
    v16h a0 = load_A_lds(sA, K, lane, k0);
#pragma unroll
    for (int t = 0; t < 4; ++t)
      acc[t] = __builtin_amdgcn_wmma_f32_16x16x32_f16(
          false, a0, false, b0[t], (short)0, acc[t], false, false);
    if (k0 + 64 < K) load_B4(Brow, k0 + 64, b0);
    v16h a1 = load_A_lds(sA, K, lane, k0 + 32);
#pragma unroll
    for (int t = 0; t < 4; ++t)
      acc[t] = __builtin_amdgcn_wmma_f32_16x16x32_f16(
          false, a1, false, b1[t], (short)0, acc[t], false, false);
  }

  size_t ob = (size_t)blockIdx.z * CC * HWP;
  int mr = m0 + ((lane >> 4) << 3);
#pragma unroll
  for (int t = 0; t < 4; ++t) {
    int nn = nb + (t << 4) + (lane & 15);
#pragma unroll
    for (int r = 0; r < 8; ++r) {
      size_t idx = ob + (size_t)(mr + r) * HWP + nn;
      out[idx] = acc[t][r] + xmid[idx];
    }
  }
}

// ---------------------------------------------------------------------------
// Host launcher
// ---------------------------------------------------------------------------
extern "C" void kernel_launch(void* const* d_in, const int* in_sizes, int n_in,
                              void* d_out, int out_size, void* d_ws, size_t ws_size,
                              hipStream_t stream) {
  (void)in_sizes; (void)n_in; (void)out_size; (void)ws_size;
  const float* x     = (const float*)d_in[0];
  const float* ln1w  = (const float*)d_in[1];
  const float* ln1b  = (const float*)d_in[2];
  const float* w1    = (const float*)d_in[3];
  const float* b1    = (const float*)d_in[4];
  const float* w3    = (const float*)d_in[5];
  const float* b3    = (const float*)d_in[6];
  const float* gnw   = (const float*)d_in[7];
  const float* gnb   = (const float*)d_in[8];
  const float* ln2w  = (const float*)d_in[9];
  const float* ln2b  = (const float*)d_in[10];
  const float* w_in  = (const float*)d_in[11];
  const float* w_dw  = (const float*)d_in[12];
  const float* w_out = (const float*)d_in[13];
  float* out = (float*)d_out;

  char* ws = (char*)d_ws;
  size_t o_y1    = 0;                                     // fp32 [B,C,HW]
  size_t o_xmid  = o_y1    + (size_t)BB * CC * HWP * 4;   // fp32 [B,C,HW]
  size_t o_y2t   = o_xmid  + (size_t)BB * CC * HWP * 4;   // f16  [B,HW,C]
  size_t o_winh  = o_y2t   + (size_t)BB * HWP * CC * 2;   // f16  [CH2,C]
  size_t o_wouth = o_winh  + (size_t)CH2 * CC * 2;        // f16  [C,KP]
  size_t o_zh    = o_wouth + (size_t)CC * KP * 2;         // f16  [B,CH2,HW]
  size_t o_zgt   = o_zh    + (size_t)BB * CH2 * HWP * 2;  // f16  [B,HW,KP]

  float*    y1    = (float*)(ws + o_y1);
  float*    xmid  = (float*)(ws + o_xmid);
  _Float16* y2t   = (_Float16*)(ws + o_y2t);
  _Float16* winh  = (_Float16*)(ws + o_winh);
  _Float16* wouth = (_Float16*)(ws + o_wouth);
  _Float16* zh    = (_Float16*)(ws + o_zh);
  _Float16* zgt   = (_Float16*)(ws + o_zgt);

  // 0) weights -> f16 (+ pad)
  k_prep<<<dim3((CH2 * CC + 255) / 256), dim3(256), 0, stream>>>(w_in, w_out, winh, wouth);
  // 1) LN1 -> y1 fp32
  k_ln<<<dim3(64, 1, BB), dim3(256), 0, stream>>>(x, ln1w, ln1b, y1, ((_Float16*)0));
  // 2) attn_ema + residual -> xmid
  k_attn<<<dim3(NG, NH, BB), dim3(256), 0, stream>>>(y1, x, w1, b1, w3, b3, gnw, gnb, xmid);
  // 3) LN2 -> y2t f16 pixel-major
  k_ln<<<dim3(64, 1, BB), dim3(256), 0, stream>>>(xmid, ln2w, ln2b, ((float*)0), y2t);
  // 4) GEMM1 (WMMA + TDM-staged A): z = W_in @ y2
  k_gemm1<<<dim3(HWP / 256, CH2 / 16, BB), dim3(128), 0, stream>>>(winh, y2t, zh);
  // 5) depthwise conv + gelu gate -> zgt pixel-major (padded K)
  k_dw<<<dim3(KP, 1, BB), dim3(256), 0, stream>>>(zh, w_dw, zgt);
  // 6) GEMM2 (WMMA + TDM-staged A) + residual -> out
  k_gemm2<<<dim3(HWP / 256, CC / 16, BB), dim3(128), 0, stream>>>(wouth, zgt, xmid, out);
}
